// MultimodalLoss_3212635537576
// MI455X (gfx1250) — compile-verified, matrix operand order checked by reading, not run on hardware
//
#include <hip/hip_runtime.h>
#include <hip/hip_bf16.h>
#include <stdint.h>

// ---------------- CDNA5 WMMA vector types (per bridge doc) ----------------
typedef __attribute__((ext_vector_type(16))) _Float16 v16h;
typedef __attribute__((ext_vector_type(8)))  float    v8f;
typedef __attribute__((ext_vector_type(4)))  float    f32x4;

// ---------------- problem constants ----------------
#define NB    8          // batch
#define NCV   1000       // num classes (logits)
#define NCLS  19         // seg classes
#define HW    (512*512)
#define DD    768
#define SMOOTHF 1e-5f
#define ALPHAF  0.25f

#define KCHUNKS (DD/32)          // 24 WMMA K-steps
#define TILE_ELEMS (KCHUNKS*32*16) // 12288 f16 per operand

// workspace layout (floats)
#define WS_FOCAL 0
#define WS_SUMX  1
#define WS_INTER (1 + NB*NCLS)
#define WS_COUNT (1 + 2*NB*NCLS)
#define WS_TOTAL 512

// seg streaming kernel geometry
constexpr int BLOCKS_PER_B  = 128;                       // 1024 blocks total
constexpr int THREADS       = 256;                       // 8 wave32
constexpr int PIX_PER_BLOCK = HW / BLOCKS_PER_B;         // 2048
constexpr int PIX_PER_ITER  = THREADS * 4;               // 1024 (float4 per thread)
constexpr int ITERS         = PIX_PER_BLOCK / PIX_PER_ITER; // 2

// ---------------- wave32 reductions ----------------
__device__ __forceinline__ float wred_sum(float v) {
#pragma unroll
  for (int off = 16; off > 0; off >>= 1) v += __shfl_xor(v, off, 32);
  return v;
}
__device__ __forceinline__ float wred_max(float v) {
#pragma unroll
  for (int off = 16; off > 0; off >>= 1) v = fmaxf(v, __shfl_xor(v, off, 32));
  return v;
}

// ---------------- kernel 0: zero workspace (harness poisons it) ----------------
__global__ void zero_ws_kernel(float* __restrict__ ws) {
  ws[threadIdx.x] = 0.0f;
}

// ---------------- kernel 1: fused dice-stats + focal streaming pass ----------------
// One pass over seg_mask [8,19,512,512]; bandwidth-bound (~167MB @ 23.3TB/s ~ 7.2us).
__global__ __launch_bounds__(THREADS)
void seg_losses_kernel(const float* __restrict__ seg,
                       const int*   __restrict__ gt,
                       float* __restrict__ ws) {
  __shared__ float s_inter[NCLS];
  __shared__ float s_count[NCLS];
  __shared__ float s_sumx [NCLS];
  __shared__ float s_focal;
  __shared__ int   s_gt[PIX_PER_ITER];   // async-DMA staging for seg_gt chunk

  const int tid = threadIdx.x;
  if (tid < NCLS) { s_inter[tid] = 0.f; s_count[tid] = 0.f; s_sumx[tid] = 0.f; }
  if (tid == 0) s_focal = 0.f;
  __syncthreads();

  const int    b    = blockIdx.x / BLOCKS_PER_B;
  const int    blk  = blockIdx.x % BLOCKS_PER_B;
  const size_t segb = (size_t)b * NCLS * HW;
  const size_t gtb  = (size_t)b * HW;

  float sumx[NCLS];
#pragma unroll
  for (int c = 0; c < NCLS; ++c) sumx[c] = 0.f;
  float focal = 0.f;

  for (int it = 0; it < ITERS; ++it) {
    const size_t p0 = (size_t)blk * PIX_PER_BLOCK + (size_t)it * PIX_PER_ITER
                    + (size_t)tid * 4;

    // CDNA5 async DMA: stage this thread's 4 gt indices into LDS (ASYNCcnt path),
    // overlapped with the 19 class-plane b128 loads below.
    {
      unsigned long long ga = (unsigned long long)(uintptr_t)(gt + gtb + p0);
      unsigned int       la = (unsigned int)(uintptr_t)(&s_gt[tid * 4]);
      asm volatile("global_load_async_to_lds_b128 %0, %1, off"
                   :: "v"(la), "v"(ga) : "memory");
    }

    // 19 coalesced, non-temporal 128-bit loads: one pixel-column in registers.
    f32x4 x[NCLS];
    const float* base = seg + segb + p0;
#pragma unroll
    for (int c = 0; c < NCLS; ++c)
      x[c] = __builtin_nontemporal_load((f32x4*)(base + (size_t)c * HW));

#pragma unroll
    for (int c = 0; c < NCLS; ++c)
      sumx[c] += (x[c][0] + x[c][1]) + (x[c][2] + x[c][3]);

    asm volatile("s_wait_asynccnt 0" ::: "memory");   // gt chunk now in LDS

#pragma unroll
    for (int q = 0; q < 4; ++q) {
      const int t = s_gt[tid * 4 + q];
      float m = -3.0e38f;
#pragma unroll
      for (int c = 0; c < NCLS; ++c) m = fmaxf(m, x[c][q]);
      float s = 0.f, xt = 0.f;
#pragma unroll
      for (int c = 0; c < NCLS; ++c) {
        const float xv = x[c][q];
        s += __expf(xv - m);
        xt = (c == t) ? xv : xt;
      }
      const float logZ  = m + __logf(s);
      const float logpt = xt - logZ;
      const float pt    = __expf(logpt);
      const float om    = 1.f - pt;
      focal += ALPHAF * om * om * (-logpt);
      atomicAdd(&s_inter[t], xt);    // ds_add_f32
      atomicAdd(&s_count[t], 1.f);
    }
  }

  // block reduction -> global atomics (58 global f32 atomics per block)
#pragma unroll
  for (int c = 0; c < NCLS; ++c) atomicAdd(&s_sumx[c], sumx[c]);
  atomicAdd(&s_focal, focal);
  __syncthreads();
  if (tid < NCLS) {
    atomicAdd(&ws[WS_SUMX  + b * NCLS + tid], s_sumx[tid]);
    atomicAdd(&ws[WS_INTER + b * NCLS + tid], s_inter[tid]);
    atomicAdd(&ws[WS_COUNT + b * NCLS + tid], s_count[tid]);
  }
  if (tid == 0) atomicAdd(&ws[WS_FOCAL], s_focal);
}

// ---------------- kernel 2: finalize (CE + modal balance + combine) ----------------
__global__ __launch_bounds__(256)
void finalize_kernel(const float* __restrict__ logits, const int* __restrict__ label,
                     const float* __restrict__ vf,     const float* __restrict__ tf,
                     const float* __restrict__ mmask,  const int* __restrict__ epoch,
                     const float* __restrict__ ws,     float* __restrict__ out) {
  // packed WMMA operand tiles, already in the 16-bit A-matrix lane layout,
  // rows 8..15 zero-padded -> wave 0 consumes with plain aligned v16h LDS loads.
  __shared__ __align__(32) _Float16 a_h[TILE_ELEMS];  // A = VN (24KB)
  __shared__ __align__(32) _Float16 b_h[TILE_ELEMS];  // B = TN^T (24KB)
  __shared__ float rnv[NB], rnt[NB];
  __shared__ float s_ce[NB];
  __shared__ float s_vc, s_tc, s_dice, s_trace;

  const int tid  = threadIdx.x;
  const int wave = tid >> 5;
  const int lane = tid & 31;
  if (tid == 0) { s_vc = 0.f; s_tc = 0.f; s_dice = 0.f; s_trace = 0.f; }
  __syncthreads();

  // ---- stage A: row norms (one wave per row) + CE (one wave per row) ----
  if (wave < NB) {
    const int r = wave;
    float av = 0.f, at = 0.f;
    for (int d = lane; d < DD; d += 32) {
      const float v = vf[r * DD + d]; av += v * v;
      const float t = tf[r * DD + d]; at += t * t;
    }
    av = wred_sum(av); at = wred_sum(at);
    if (lane == 0) { rnv[r] = rsqrtf(av); rnt[r] = rsqrtf(at); }
  }
  if (wave < NB) {
    const int r = wave;
    float m = -3.0e38f;
    for (int i = lane; i < NCV; i += 32) m = fmaxf(m, logits[r * NCV + i]);
    m = wred_max(m);
    float s = 0.f;
    for (int i = lane; i < NCV; i += 32) s += __expf(logits[r * NCV + i] - m);
    s = wred_sum(s);
    if (lane == 0) {
      const int lb = label[r];
      s_ce[r] = (m + __logf(s)) - logits[r * NCV + lb];
    }
  }
  __syncthreads();

  // ---- stage B1: pack WMMA operand tiles (branch cost irrelevant here) ----
  // element e -> (chunk, lane2, j); lane2=(m,hi); K = chunk*32 + j + hi*8 + (j>=8)*8
  for (int e = tid; e < TILE_ELEMS; e += 256) {
    const int chunk = e >> 9;          // /512
    const int rem   = e & 511;
    const int lane2 = rem >> 4;
    const int j     = rem & 15;
    const int mrow  = lane2 & 15;
    const int hi    = lane2 >> 4;
    const int K     = chunk * 32 + j + hi * 8 + ((j >= 8) ? 8 : 0);
    _Float16 av = (_Float16)0.f, bv = (_Float16)0.f;
    if (mrow < NB) {
      av = (_Float16)(vf[mrow * DD + K] * rnv[mrow]);
      bv = (_Float16)(tf[mrow * DD + K] * rnt[mrow]);
    }
    a_h[e] = av;
    b_h[e] = bv;
  }

  // ---- stage B2: consistency terms mean((vn - colmean)^2), exact f32 ----
  {
    float vc = 0.f, tc = 0.f;
    for (int d = tid; d < DD; d += 256) {
      float vv[NB], tv[NB];
      float sv = 0.f, st = 0.f;
#pragma unroll
      for (int bb = 0; bb < NB; ++bb) {
        vv[bb] = vf[bb * DD + d] * rnv[bb]; sv += vv[bb];
        tv[bb] = tf[bb * DD + d] * rnt[bb]; st += tv[bb];
      }
      const float mv = sv * (1.0f / NB), mt = st * (1.0f / NB);
#pragma unroll
      for (int bb = 0; bb < NB; ++bb) {
        const float e1 = vv[bb] - mv; vc += e1 * e1;
        const float e2 = tv[bb] - mt; tc += e2 * e2;
      }
    }
    atomicAdd(&s_vc, vc);
    atomicAdd(&s_tc, tc);
  }
  __syncthreads();

  // ---- stage C1: cross term via WMMA chain: trace(VN * TN^T), K=768 ----
  // Branch-free feed: 2x ds_load_b128 per operand per chunk, fully unrolled.
  if (wave == 0) {
    const v16h* ap = (const v16h*)a_h;
    const v16h* bp = (const v16h*)b_h;
    v8f acc = {};
#pragma unroll
    for (int ch = 0; ch < KCHUNKS; ++ch) {
      const v16h a  = ap[ch * 32 + lane];
      const v16h bm = bp[ch * 32 + lane];
      acc = __builtin_amdgcn_wmma_f32_16x16x32_f16(
          /*neg_a=*/false, a, /*neg_b=*/false, bm,
          /*c_mod=*/(short)0, acc, /*reuse_a=*/false, /*reuse_b=*/false);
    }
    // diagonal: lane l<8 holds D[l][l] in acc element l (C/D layout table)
    float dg = 0.f;
#pragma unroll
    for (int v = 0; v < 8; ++v) dg = (lane == v) ? acc[v] : dg;
    dg = wred_sum(dg);
    if (lane == 0) s_trace = dg;
  }

  // ---- stage C2: dice from workspace statistics ----
  if (tid < NB * NCLS) {
    const float inter = ws[WS_INTER + tid];
    const float sx    = ws[WS_SUMX  + tid];
    const float cnt   = ws[WS_COUNT + tid];
    const float dice  = (2.f * inter + SMOOTHF) / (sx + cnt + SMOOTHF);
    atomicAdd(&s_dice, 1.f - dice);
  }
  __syncthreads();

  // ---- combine ----
  if (tid == 0) {
    float ce = 0.f;
#pragma unroll
    for (int r = 0; r < NB; ++r) ce += s_ce[r];
    ce *= (1.0f / NB);
    const float v_cons = s_vc * (1.0f / (NB * DD));
    const float t_cons = s_tc * (1.0f / (NB * DD));
    const float cross  = 1.f - s_trace * (1.0f / NB);
    float mm0 = 0.f, mm1 = 0.f;
#pragma unroll
    for (int r = 0; r < NB; ++r) { mm0 += mmask[2 * r]; mm1 += mmask[2 * r + 1]; }
    mm0 *= (1.0f / NB); mm1 *= (1.0f / NB);
    const float beta      = 0.5f * __powf(0.99f, (float)epoch[0]);
    const float mb        = (1.f - beta) * v_cons * mm0 + beta * t_cons * mm1 + cross;
    const float focal     = ws[WS_FOCAL] * (1.0f / ((float)NB * (float)HW));
    const float dice_loss = s_dice * (1.0f / (NB * NCLS));
    out[0] = ce + 0.3f * mb + 0.5f * (dice_loss + focal);
  }
}

// ---------------- launcher ----------------
extern "C" void kernel_launch(void* const* d_in, const int* in_sizes, int n_in,
                              void* d_out, int out_size, void* d_ws, size_t ws_size,
                              hipStream_t stream) {
  const float* logits = (const float*)d_in[0];
  const int*   label  = (const int*)  d_in[1];
  const float* vfeat  = (const float*)d_in[2];
  const float* tfeat  = (const float*)d_in[3];
  const float* mmask  = (const float*)d_in[4];
  const float* seg    = (const float*)d_in[5];
  const int*   gt     = (const int*)  d_in[6];
  const int*   epoch  = (const int*)  d_in[7];
  float*       out    = (float*)d_out;
  float*       ws     = (float*)d_ws;

  zero_ws_kernel<<<1, WS_TOTAL, 0, stream>>>(ws);
  seg_losses_kernel<<<dim3(NB * BLOCKS_PER_B), dim3(THREADS), 0, stream>>>(seg, gt, ws);
  finalize_kernel<<<1, 256, 0, stream>>>(logits, label, vfeat, tfeat, mmask, epoch, ws, out);
}
